// SRUCell_53996328845368
// MI455X (gfx1250) — compile-verified, matrix operand order checked by reading, not run on hardware
//
#include <hip/hip_runtime.h>
#include <hip/hip_bf16.h>
#include <math.h>

typedef __attribute__((ext_vector_type(16))) _Float16 v16h;
typedef __attribute__((ext_vector_type(8)))  float    v8f;

union Frag {
    v16h  v;
    uint4 q[2];
};
union Pack4 {
    _Float16 h[4];
    uint2    u;
};

#define TILE_M 128
#define TILE_N 64
#define TILE_K 32
#define LDA_PAD 40   // 80B rows: 16B-aligned b128 frag loads, conflict-free (20*row % 64 distinct)
#define LDB_PAD 40
#define NCHUNK 32

__device__ __forceinline__ float fast_sigmoid(float x) {
    return __builtin_amdgcn_rcpf(1.0f + __expf(-x));
}
__device__ __forceinline__ float fast_tanh(float x) {
    // tanh(x) = 1 - 2/(exp(2x)+1); exp->inf => 1, exp->0 => -1
    return 1.0f - 2.0f * __builtin_amdgcn_rcpf(__expf(2.0f * x) + 1.0f);
}

// ---------------------------------------------------------------------------
// GEMM: U[M,3d] = X[M,d] * W[d,3d], fused gate epilogue.
//   col n: i = n/3, which = n%3
//   which==0 -> xt_out[m,i]  (fp32, lives in d_out h-region)
//   which==1 -> ft_out[m,i]  (f16, sigmoid(u + bias[i]))
//   which==2 -> r_out [m,i]  (f16, sigmoid(u + bias[d+i]))
// Double-buffered LDS + register prefetch; 2x2 WMMA tiles per wave.
// ---------------------------------------------------------------------------
__global__ __launch_bounds__(256) void sru_gemm_wmma(
    const float* __restrict__ X, const float* __restrict__ W,
    const float* __restrict__ bias,
    float* __restrict__ xt_out, _Float16* __restrict__ ft_out,
    _Float16* __restrict__ r_out, int d)
{
    const int K = d;
    const int N = 3 * d;

    __shared__ _Float16 As[2][TILE_M * LDA_PAD];
    __shared__ _Float16 Bs[2][TILE_N * LDB_PAD];

    const int tid      = threadIdx.x;
    const int lane     = tid & 31;
    const int wave     = tid >> 5;
    const int laneHalf = lane >> 4;        // 0: lanes 0-15, 1: lanes 16-31
    const int laneRow  = lane & 15;
    const int waveM    = (wave & 3) * 32;  // 4 waves along M
    const int waveN    = (wave >> 2) * 32; // 2 waves along N
    const int blockM   = blockIdx.y * TILE_M;
    const int blockN   = blockIdx.x * TILE_N;

    // cooperative-load coordinates (fixed per thread)
    const int aRow[4] = { (tid + 0)   >> 3, (tid + 256) >> 3,
                          (tid + 512) >> 3, (tid + 768) >> 3 };
    const int aK4     = (tid & 7) << 2;
    const int bK [2]  = { (tid + 0) >> 4, (tid + 256) >> 4 };
    const int bN4     = (tid & 15) << 2;

    float4 aReg[4];
    float4 bReg[2];

    auto loadTiles = [&](int kc) {
#pragma unroll
        for (int s = 0; s < 4; ++s)
            aReg[s] = *(const float4*)&X[(size_t)(blockM + aRow[s]) * K + kc + aK4];
#pragma unroll
        for (int s = 0; s < 2; ++s)
            bReg[s] = *(const float4*)&W[(size_t)(kc + bK[s]) * N + blockN + bN4];
    };
    auto storeTiles = [&](int buf) {
#pragma unroll
        for (int s = 0; s < 4; ++s) {
            Pack4 p;
            p.h[0] = (_Float16)aReg[s].x; p.h[1] = (_Float16)aReg[s].y;
            p.h[2] = (_Float16)aReg[s].z; p.h[3] = (_Float16)aReg[s].w;
            *(uint2*)&As[buf][aRow[s] * LDA_PAD + aK4] = p.u;   // 8B aligned
        }
#pragma unroll
        for (int s = 0; s < 2; ++s) {   // transpose into Bs[n][k]
            Bs[buf][(bN4 + 0) * LDB_PAD + bK[s]] = (_Float16)bReg[s].x;
            Bs[buf][(bN4 + 1) * LDB_PAD + bK[s]] = (_Float16)bReg[s].y;
            Bs[buf][(bN4 + 2) * LDB_PAD + bK[s]] = (_Float16)bReg[s].z;
            Bs[buf][(bN4 + 3) * LDB_PAD + bK[s]] = (_Float16)bReg[s].w;
        }
    };

    v8f acc[2][2] = {};

    loadTiles(0);
    storeTiles(0);
    __syncthreads();

    int buf = 0;
    for (int kc = 0; kc < K; kc += TILE_K) {
        const bool hasNext = (kc + TILE_K) < K;
        if (hasNext) loadTiles(kc + TILE_K);   // overlap global loads with WMMA

        // frags: lane<16 holds K {0..7,16..23}; lane>=16 holds K {8..15,24..31}
        // -> each fragment is two contiguous 16B LDS reads.
        Frag a[2], b[2];
#pragma unroll
        for (int t = 0; t < 2; ++t) {
            const _Float16* pa = &As[buf][(waveM + t * 16 + laneRow) * LDA_PAD + (laneHalf << 3)];
            a[t].q[0] = *(const uint4*)pa;
            a[t].q[1] = *(const uint4*)(pa + 16);
            const _Float16* pb = &Bs[buf][(waveN + t * 16 + laneRow) * LDB_PAD + (laneHalf << 3)];
            b[t].q[0] = *(const uint4*)pb;
            b[t].q[1] = *(const uint4*)(pb + 16);
        }
#pragma unroll
        for (int tm = 0; tm < 2; ++tm)
#pragma unroll
            for (int tn = 0; tn < 2; ++tn)
                acc[tm][tn] = __builtin_amdgcn_wmma_f32_16x16x32_f16(
                    false, a[tm].v, false, b[tn].v,
                    (short)0, acc[tm][tn], false, false);

        if (hasNext) {
            storeTiles(buf ^ 1);
            __syncthreads();
            buf ^= 1;
        }
    }

    // ---- epilogue: D layout -> M = v + 8*laneHalf, N = lane&15 ----
#pragma unroll
    for (int tm = 0; tm < 2; ++tm) {
#pragma unroll
        for (int tn = 0; tn < 2; ++tn) {
#pragma unroll
            for (int v = 0; v < 8; ++v) {
                int m    = blockM + waveM + tm * 16 + v + (laneHalf << 3);
                int ncol = blockN + waveN + tn * 16 + laneRow;
                float val = acc[tm][tn][v];
                int i = ncol / 3;
                int which = ncol - i * 3;
                size_t off = (size_t)m * d + i;
                if (which == 0) {
                    xt_out[off] = val;
                } else if (which == 1) {
                    ft_out[off] = (_Float16)fast_sigmoid(val + bias[i]);
                } else {
                    r_out[off] = (_Float16)fast_sigmoid(val + bias[d + i]);
                }
            }
        }
    }
}

// ---------------------------------------------------------------------------
// Phase A: per (chunk, channel) reduce chunk to affine c_out = A*c_in + B
//   step: c = ft*c + (1-ft)*xt
// ---------------------------------------------------------------------------
__global__ __launch_bounds__(256) void sru_scan_reduce(
    const _Float16* __restrict__ ft_buf, const float* __restrict__ xt_buf,
    float* __restrict__ chunkA, float* __restrict__ chunkB,
    int CH, int clen)
{
    int tid   = blockIdx.x * blockDim.x + threadIdx.x;
    int ch    = tid & (CH - 1);
    int chunk = tid / CH;
    size_t base = (size_t)chunk * clen * CH + ch;

    float Aacc = 1.0f, Bacc = 0.0f;
#pragma unroll 8
    for (int j = 0; j < clen; ++j) {
        size_t idx = base + (size_t)j * CH;
        float ft = (float)ft_buf[idx];
        float xt = xt_buf[idx];
        Aacc *= ft;
        Bacc = Bacc * ft + xt * (1.0f - ft);
    }
    chunkA[(size_t)chunk * CH + ch] = Aacc;
    chunkB[(size_t)chunk * CH + ch] = Bacc;
}

// ---------------------------------------------------------------------------
// Phase B: stitch chunk carries; emit carry-in per chunk and c_last.
// ---------------------------------------------------------------------------
__global__ __launch_bounds__(256) void sru_scan_stitch(
    const float* __restrict__ chunkA, const float* __restrict__ chunkB,
    const float* __restrict__ c0, float* __restrict__ carry,
    float* __restrict__ c_last, int CH, int nchunk)
{
    int ch = blockIdx.x * blockDim.x + threadIdx.x;
    float c = c0[ch];
    for (int j = 0; j < nchunk; ++j) {
        size_t idx = (size_t)j * CH + ch;
        carry[idx] = c;
        c = chunkA[idx] * c + chunkB[idx];
    }
    c_last[ch] = c;
}

// ---------------------------------------------------------------------------
// Phase C: replay chunk with known carry; h = (tanh(c) - x)*r + x.
// xt lives in d_out's h-region, overwritten in place (read-before-write).
// ---------------------------------------------------------------------------
__global__ __launch_bounds__(256) void sru_scan_replay(
    const _Float16* __restrict__ ft_buf, const _Float16* __restrict__ r_buf,
    const float* __restrict__ carry, const float* __restrict__ x,
    float* __restrict__ hxt, int CH, int clen)
{
    int tid   = blockIdx.x * blockDim.x + threadIdx.x;
    int ch    = tid & (CH - 1);
    int chunk = tid / CH;
    size_t base = (size_t)chunk * clen * CH + ch;

    float c = carry[(size_t)chunk * CH + ch];
#pragma unroll 8
    for (int j = 0; j < clen; ++j) {
        size_t idx = base + (size_t)j * CH;
        float ft = (float)ft_buf[idx];
        float xt = hxt[idx];
        c = (c - xt) * ft + xt;
        float g  = fast_tanh(c);
        float rv = (float)r_buf[idx];
        float xv = x[idx];
        hxt[idx] = (g - xv) * rv + xv;
    }
}

extern "C" void kernel_launch(void* const* d_in, const int* in_sizes, int n_in,
                              void* d_out, int out_size, void* d_ws, size_t ws_size,
                              hipStream_t stream)
{
    const float* X    = (const float*)d_in[0];  // [L,B,d]
    const float* W    = (const float*)d_in[1];  // [d,3d]
    const float* bias = (const float*)d_in[2];  // [2d]
    const float* c0   = (const float*)d_in[3];  // [B,d]

    const int d  = in_sizes[2] / 2;
    const int Bb = in_sizes[3] / d;
    const int L  = in_sizes[0] / (Bb * d);
    const int M  = L * Bb;           // 32768
    const int N  = 3 * d;            // 3072
    const int CH = Bb * d;           // 16384 channels
    const int clen = L / NCHUNK;     // 64

    float* h_out   = (float*)d_out;            // [M, d] -> xt then h (in place)
    float* c_last  = h_out + (size_t)M * d;    // [CH]

    char* ws = (char*)d_ws;
    _Float16* ft_buf = (_Float16*)ws;                                    // M*d halves
    _Float16* r_buf  = (_Float16*)(ws + (size_t)M * d * 2);              // M*d halves
    float* chunkA    = (float*)(ws + (size_t)M * d * 4);                 // NCHUNK*CH
    float* chunkB    = chunkA + (size_t)NCHUNK * CH;
    float* carry     = chunkB + (size_t)NCHUNK * CH;

    // 1) GEMM + gate epilogue (WMMA f16 -> f32)
    dim3 ggrid(N / TILE_N, M / TILE_M);
    sru_gemm_wmma<<<ggrid, 256, 0, stream>>>(X, W, bias, h_out, ft_buf, r_buf, d);

    // 2) chunked parallel scan
    int totalAC = NCHUNK * CH;
    sru_scan_reduce<<<totalAC / 256, 256, 0, stream>>>(ft_buf, h_out, chunkA, chunkB, CH, clen);
    sru_scan_stitch<<<CH / 256, 256, 0, stream>>>(chunkA, chunkB, c0, carry, c_last, CH, NCHUNK);
    sru_scan_replay<<<totalAC / 256, 256, 0, stream>>>(ft_buf, r_buf, carry, X, h_out, CH, clen);
}
